// TExpert_74277164417701
// MI455X (gfx1250) — matrix operand order repaired and checked by reading, not compiled, not run on hardware
//
#include <hip/hip_runtime.h>
#include <hip/hip_bf16.h>

// ---------------------------------------------------------------------------
// Dense MoE (2 steps) for MI455X / gfx1250, wave32 + WMMA f16 path.
// B=2, S=2048 -> T=4096 tokens, H=1024, I=4096, E=8.
// GEMM: 128x128 WG tile, K chunked by 64, double-buffered LDS A staging
// (one workgroup barrier per 64-K chunk, 16 v_wmma per chunk per wave).
// ---------------------------------------------------------------------------

#define T_TOK 4096
#define H_DIM 1024
#define I_DIM 4096
#define E_EXP 8
#define RMS_EPS 1e-6f

typedef __attribute__((ext_vector_type(16))) _Float16 v16h;
typedef __attribute__((ext_vector_type(8)))  _Float16 v8h;
typedef __attribute__((ext_vector_type(8)))  float    v8f;

union HFrag {
    v16h f;
    v8h  h[2];
};

// ---------------------------------------------------------------------------
// Transpose + convert fp32 [R][C] (per expert, blockIdx.z) -> f16 [C][R].
// Makes WMMA B-fragment loads contiguous (16 K-values per lane).
// ---------------------------------------------------------------------------
__global__ void transpose_f32_to_f16(const float* __restrict__ in,
                                     _Float16* __restrict__ out,
                                     int R, int C) {
    __shared__ float tile[32][33];
    const size_t ebase = (size_t)blockIdx.z * (size_t)R * (size_t)C;
    const int c0 = blockIdx.x * 32;
    const int r0 = blockIdx.y * 32;
    const int tx = threadIdx.x & 31;
    const int ty = threadIdx.x >> 5;  // 0..7 (256 threads)

#pragma unroll
    for (int j = ty; j < 32; j += 8)
        tile[j][tx] = in[ebase + (size_t)(r0 + j) * C + c0 + tx];
    __syncthreads();
#pragma unroll
    for (int j = ty; j < 32; j += 8)
        out[ebase + (size_t)(c0 + j) * R + r0 + tx] = (_Float16)tile[tx][j];
}

// ---------------------------------------------------------------------------
// Per-token prep: t16 = (use_norm ? rmsnorm(x, ln_w) : x) as f16,
// comb = x (accumulator for x + sum_e rw*eo). One block per token.
// ---------------------------------------------------------------------------
__global__ void prep_kernel(const float* __restrict__ x,
                            const float* __restrict__ lnw,
                            _Float16* __restrict__ t16,
                            float* __restrict__ comb,
                            int use_norm) {
    const int token = blockIdx.x;
    const float* xr = x + (size_t)token * H_DIM;

    float ss = 0.f;
    for (int h = threadIdx.x; h < H_DIM; h += 256) {
        float v = xr[h];
        ss += v * v;
    }
    __shared__ float red[8];
    __shared__ float stot;
#pragma unroll
    for (int off = 16; off > 0; off >>= 1) ss += __shfl_down(ss, off);
    if ((threadIdx.x & 31) == 0) red[threadIdx.x >> 5] = ss;
    __syncthreads();
    if (threadIdx.x == 0) {
        float t = 0.f;
#pragma unroll
        for (int i = 0; i < 8; ++i) t += red[i];
        stot = t;
    }
    __syncthreads();
    const float scale = use_norm ? rsqrtf(stot * (1.f / H_DIM) + RMS_EPS) : 1.f;

    for (int h = threadIdx.x; h < H_DIM; h += 256) {
        float v  = xr[h];
        float tv = use_norm ? v * scale * lnw[h] : v;
        t16[(size_t)token * H_DIM + h]  = (_Float16)tv;
        comb[(size_t)token * H_DIM + h] = v;
    }
}

// ---------------------------------------------------------------------------
// Gate logits + softmax: one wave (32 lanes) per token, 8 tokens per block.
// rw[t][e] = softmax_e( t16[t] . gate_w[e] )
// ---------------------------------------------------------------------------
__global__ void gate_softmax_kernel(const _Float16* __restrict__ t16,
                                    const float* __restrict__ gw,
                                    float* __restrict__ rw) {
    const int token = blockIdx.x * 8 + (threadIdx.x >> 5);
    const int lane  = threadIdx.x & 31;

    float acc[E_EXP];
#pragma unroll
    for (int e = 0; e < E_EXP; ++e) acc[e] = 0.f;

    for (int h = lane; h < H_DIM; h += 32) {
        const float tv = (float)t16[(size_t)token * H_DIM + h];
#pragma unroll
        for (int e = 0; e < E_EXP; ++e)
            acc[e] += tv * gw[(size_t)e * H_DIM + h];
    }
#pragma unroll
    for (int off = 16; off > 0; off >>= 1) {
#pragma unroll
        for (int e = 0; e < E_EXP; ++e) acc[e] += __shfl_down(acc[e], off);
    }
    if (lane == 0) {
        float mx = acc[0];
#pragma unroll
        for (int e = 1; e < E_EXP; ++e) mx = fmaxf(mx, acc[e]);
        float s = 0.f;
        float ex[E_EXP];
#pragma unroll
        for (int e = 0; e < E_EXP; ++e) { ex[e] = __expf(acc[e] - mx); s += ex[e]; }
        const float inv = 1.f / s;
#pragma unroll
        for (int e = 0; e < E_EXP; ++e) rw[(size_t)token * E_EXP + e] = ex[e] * inv;
    }
}

// ---------------------------------------------------------------------------
// 128x128-tile f16 WMMA GEMM.
//   C_tile = A[M,K] (row major, f16) x Bt[N,K]^T (per-column contiguous f16)
// 256 threads = 8 waves; wave grid 4(M) x 2(N); each wave owns 2 M-tiles x
// 4 N-tiles of 16x16. K chunked by 64 with double-buffered LDS A staging:
// one barrier + 16 v_wmma_f32_16x16x32_f16 per chunk per wave.
// MODE 0: out = f16( silu(acc + bias) )          (gemm1: h = silu(t@w1+b1))
// MODE 1: comb += rw[row][e] * (acc + bias)      (gemm2 fused combine)
// ---------------------------------------------------------------------------
template <int MODE>
__global__ void __launch_bounds__(256)
moe_gemm128(const _Float16* __restrict__ A,
            const _Float16* __restrict__ Bt,
            const float* __restrict__ bias,
            const float* __restrict__ rw,
            void* __restrict__ Cout,
            int K, int N, int expert) {
    __shared__ _Float16 ldsA[2][128 * 64];  // 2 x 16 KB ping-pong

    const int tid    = threadIdx.x;
    const int lane   = tid & 31;
    const int wid    = tid >> 5;
    const int wave_m = wid & 3;   // 0..3 -> 32-row slab
    const int wave_n = wid >> 2;  // 0..1 -> 64-col slab
    const int blockN = blockIdx.x * 128;
    const int blockM = blockIdx.y * 128;

    const int mrow  = lane & 15;  // also the N index inside a C tile
    const int khalf = lane >> 4;  // 0 or 1
    const int kh    = khalf * 16;

    // staging assignment: each thread moves 2 rows x 16 contiguous elems
    const int st_row = tid >> 2;        // 0..63 (and +64)
    const int st_off = (tid & 3) * 16;  // 0/16/32/48

    v8f acc[2][4];
#pragma unroll
    for (int mi = 0; mi < 2; ++mi)
#pragma unroll
        for (int ni = 0; ni < 4; ++ni)
#pragma unroll
            for (int r = 0; r < 8; ++r) acc[mi][ni][r] = 0.f;

    const int nchunks = K >> 6;  // K / 64

    // ---- prologue: stage chunk 0 into buffer 0
    {
        const _Float16* ap0 = A + (size_t)(blockM + st_row) * K + st_off;
        const _Float16* ap1 = A + (size_t)(blockM + st_row + 64) * K + st_off;
        v8h r0a = *(const v8h*)ap0;
        v8h r0b = *(const v8h*)(ap0 + 8);
        v8h r1a = *(const v8h*)ap1;
        v8h r1b = *(const v8h*)(ap1 + 8);
        *(v8h*)&ldsA[0][st_row * 64 + st_off]            = r0a;
        *(v8h*)&ldsA[0][st_row * 64 + st_off + 8]        = r0b;
        *(v8h*)&ldsA[0][(st_row + 64) * 64 + st_off]     = r1a;
        *(v8h*)&ldsA[0][(st_row + 64) * 64 + st_off + 8] = r1b;
    }
    __syncthreads();

    for (int c = 0; c < nchunks; ++c) {
        const int k0 = c << 6;
        const _Float16* cur = ldsA[c & 1];
        const bool have_next = (c + 1) < nchunks;

        // ---- issue global loads for next chunk early (into registers)
        v8h n0a, n0b, n1a, n1b;
        if (have_next) {
            const _Float16* ap0 =
                A + (size_t)(blockM + st_row) * K + k0 + 64 + st_off;
            const _Float16* ap1 =
                A + (size_t)(blockM + st_row + 64) * K + k0 + 64 + st_off;
            n0a = *(const v8h*)ap0;
            n0b = *(const v8h*)(ap0 + 8);
            n1a = *(const v8h*)ap1;
            n1b = *(const v8h*)(ap1 + 8);
            // global_prefetch_b8 of the next chunk's B stream
            __builtin_prefetch(
                Bt + (size_t)(blockN + wave_n * 64 + mrow) * K + k0 + 64 + kh, 0, 1);
        }

        // ---- consume current chunk: two 32-K sub-chunks, 8 WMMA each
#pragma unroll
        for (int s = 0; s < 2; ++s) {
            const int ks = s * 32;

            // A fragments from LDS (16-bit A layout: lanes 0-15 hold
            // K 0-7 / 16-23, lanes 16-31 hold K 8-15 / 24-31)
            HFrag a[2];
#pragma unroll
            for (int mi = 0; mi < 2; ++mi) {
                const int rr = wave_m * 32 + mi * 16 + mrow;
                a[mi].h[0] = *(const v8h*)&cur[rr * 64 + ks + khalf * 8];
                a[mi].h[1] = *(const v8h*)&cur[rr * 64 + ks + 16 + khalf * 8];
            }

            // B fragments direct from L2 (pre-transposed weights: lane =
            // output column, K 8h..8h+7 and 16+8h..16+8h+7 -> two b128 loads)
            HFrag b[4];
#pragma unroll
            for (int ni = 0; ni < 4; ++ni) {
                const _Float16* bp =
                    Bt + (size_t)(blockN + wave_n * 64 + ni * 16 + mrow) * K +
                    k0 + ks + khalf * 8;
                b[ni].h[0] = *(const v8h*)bp;
                b[ni].h[1] = *(const v8h*)(bp + 16);
            }

#pragma unroll
            for (int mi = 0; mi < 2; ++mi)
#pragma unroll
                for (int ni = 0; ni < 4; ++ni)
                    acc[mi][ni] = __builtin_amdgcn_wmma_f32_16x16x32_f16(
                        false, a[mi].f, false, b[ni].f,
                        (short)0, acc[mi][ni], false, false);
        }

        // ---- store next chunk into the other buffer; single barrier
        if (have_next) {
            _Float16* nxt = (_Float16*)ldsA[(c + 1) & 1];
            *(v8h*)&nxt[st_row * 64 + st_off]            = n0a;
            *(v8h*)&nxt[st_row * 64 + st_off + 8]        = n0b;
            *(v8h*)&nxt[(st_row + 64) * 64 + st_off]     = n1a;
            *(v8h*)&nxt[(st_row + 64) * 64 + st_off + 8] = n1b;
            __syncthreads();
        }
    }

    // ---- epilogue (C layout: VGPR r, lanes 0-15 -> M=r, N=lane;
    //      lanes 16-31 -> M=8+r)
#pragma unroll
    for (int mi = 0; mi < 2; ++mi) {
#pragma unroll
        for (int ni = 0; ni < 4; ++ni) {
            const int row0 = blockM + wave_m * 32 + mi * 16 + khalf * 8;
            const int col  = blockN + wave_n * 64 + ni * 16 + mrow;
            const float bv = bias[col];
            if (MODE == 0) {
                _Float16* out = (_Float16*)Cout;
#pragma unroll
                for (int r = 0; r < 8; ++r) {
                    float v = acc[mi][ni][r] + bv;
                    v = v * (1.f / (1.f + __expf(-v)));  // silu
                    out[(size_t)(row0 + r) * N + col] = (_Float16)v;
                }
            } else {
                float* out = (float*)Cout;
#pragma unroll
                for (int r = 0; r < 8; ++r) {
                    const int row = row0 + r;
                    const float w = rw[(size_t)row * E_EXP + expert];
                    const float v = (acc[mi][ni][r] + bv) * w;
                    out[(size_t)row * N + col] += v;  // disjoint tiles: race-free
                }
            }
        }
    }
}

__global__ void copy_f32_kernel(const float* __restrict__ src,
                                float* __restrict__ dst, int n) {
    const int i = blockIdx.x * 256 + threadIdx.x;
    if (i < n) dst[i] = src[i];
}

// ---------------------------------------------------------------------------
// Orchestration
// ---------------------------------------------------------------------------
extern "C" void kernel_launch(void* const* d_in, const int* in_sizes, int n_in,
                              void* d_out, int out_size, void* d_ws, size_t ws_size,
                              hipStream_t stream) {
    const float* x_in   = (const float*)d_in[0];  // [B,S,H]
    const float* gate_w = (const float*)d_in[1];  // [E,H]
    const float* w1     = (const float*)d_in[2];  // [E,H,I]
    const float* b1     = (const float*)d_in[3];  // [E,I]
    const float* w2     = (const float*)d_in[4];  // [E,I,H]
    const float* b2     = (const float*)d_in[5];  // [E,H]
    const float* ln_w   = (const float*)d_in[6];  // [H]
    float* out          = (float*)d_out;

    char* ws = (char*)d_ws;
    _Float16* w1t = (_Float16*)ws; ws += (size_t)E_EXP * H_DIM * I_DIM * 2;  // [E][I][H]
    _Float16* w2t = (_Float16*)ws; ws += (size_t)E_EXP * I_DIM * H_DIM * 2;  // [E][H][I]
    _Float16* t16 = (_Float16*)ws; ws += (size_t)T_TOK * H_DIM * 2;          // [T][H]
    _Float16* h16 = (_Float16*)ws; ws += (size_t)T_TOK * I_DIM * 2;          // [T][I]
    float* rw     = (float*)ws;    ws += (size_t)T_TOK * E_EXP * 4;          // [T][E]
    float* combA  = (float*)ws;    ws += (size_t)T_TOK * H_DIM * 4;          // [T][H]
    float* combB  = (float*)ws;    ws += (size_t)T_TOK * H_DIM * 4;          // [T][H]

    const dim3 blk(256);

    // One-shot (per call) weight convert + transpose: fp32 HBM -> f16 (fits L2)
    transpose_f32_to_f16<<<dim3(I_DIM / 32, H_DIM / 32, E_EXP), blk, 0, stream>>>(
        w1, w1t, H_DIM, I_DIM);
    transpose_f32_to_f16<<<dim3(H_DIM / 32, I_DIM / 32, E_EXP), blk, 0, stream>>>(
        w2, w2t, I_DIM, H_DIM);

    const float* xcur = x_in;
    float* combs[2] = {combA, combB};
    for (int step = 0; step < 2; ++step) {
        float* comb = combs[step];
        prep_kernel<<<T_TOK, blk, 0, stream>>>(xcur, ln_w, t16, comb, step > 0 ? 1 : 0);
        gate_softmax_kernel<<<T_TOK / 8, blk, 0, stream>>>(t16, gate_w, rw);
        for (int e = 0; e < E_EXP; ++e) {
            // h = silu(t @ w1[e] + b1[e])  : M=T, K=H, N=I
            moe_gemm128<0><<<dim3(I_DIM / 128, T_TOK / 128), blk, 0, stream>>>(
                t16, w1t + (size_t)e * I_DIM * H_DIM, b1 + (size_t)e * I_DIM,
                nullptr, (void*)h16, H_DIM, I_DIM, e);
            // comb += rw[:,e] * (h @ w2[e] + b2[e]) : M=T, K=I, N=H
            moe_gemm128<1><<<dim3(H_DIM / 128, T_TOK / 128), blk, 0, stream>>>(
                h16, w2t + (size_t)e * H_DIM * I_DIM, b2 + (size_t)e * H_DIM,
                rw, (void*)comb, I_DIM, H_DIM, e);
        }
        xcur = comb;
    }

    copy_f32_kernel<<<(T_TOK * H_DIM + 255) / 256, blk, 0, stream>>>(
        combs[1], out, T_TOK * H_DIM);
}
